// PositionPassingTGN_50010599194850
// MI455X (gfx1250) — matrix operand order repaired and checked by reading, not compiled, not run on hardware
//
#include <hip/hip_runtime.h>
#include <hip/hip_bf16.h>
#include <math.h>

// Problem constants (match reference)
#define NN  65536
#define D   128
#define KIN 512          // message width
#define TM  64           // rows (nodes) per block -> halves L2 weight re-fetch
#define NMT (TM / 16)    // 16-row WMMA sub-tiles per block
#define SA_STRIDE (KIN + 8)   // halves; 1040B row stride keeps 16B-aligned b128 LDS loads

// f16-weight workspace layout (in halves)
#define OFF_WIH_MEM 0
#define OFF_WHH_MEM 196608                 // 384*512
#define OFF_WIH_POS 245760                 // + 384*128
#define OFF_WHH_POS 442368                 // + 384*512
#define W_TOTAL     491520                 // + 384*128

typedef _Float16 half_t;
typedef __attribute__((ext_vector_type(16))) _Float16 v16h;
typedef __attribute__((ext_vector_type(8)))  _Float16 v8h;
typedef __attribute__((ext_vector_type(8)))  float    v8f;

union V16U { v16h v; v8h h[2]; };

// B fragment: 16 consecutive f16 (32B) per lane. lane&15 = N, lane>>4 selects K half.
__device__ __forceinline__ v16h load_bfrag(const half_t* p) {
  V16U u;
  u.h[0] = *(const v8h*)(p);
  u.h[1] = *(const v8h*)(p + 8);
  return u.v;
}

// A fragment from LDS row: halves[0..15] = K kbase(+16 if lane>=16).. contiguous
__device__ __forceinline__ v16h load_afrag(const half_t* sa_row, int kbase, int laneHi) {
  V16U u;
  u.h[0] = *(const v8h*)(sa_row + kbase + laneHi * 16);
  u.h[1] = *(const v8h*)(sa_row + kbase + laneHi * 16 + 8);
  return u.v;
}

// ---------------------------------------------------------------------------
// Prologue: convert the four GRU weight matrices to f16 in workspace.
// ---------------------------------------------------------------------------
__global__ __launch_bounds__(256) void convert_weights_f16(
    const float* __restrict__ Wih_m, const float* __restrict__ Whh_m,
    const float* __restrict__ Wih_p, const float* __restrict__ Whh_p,
    half_t* __restrict__ ws) {
  int i = blockIdx.x * blockDim.x + threadIdx.x;
  if (i >= W_TOTAL) return;
  float v;
  if      (i < OFF_WHH_MEM) v = Wih_m[i - OFF_WIH_MEM];
  else if (i < OFF_WIH_POS) v = Whh_m[i - OFF_WHH_MEM];
  else if (i < OFF_WHH_POS) v = Wih_p[i - OFF_WIH_POS];
  else                      v = Whh_p[i - OFF_WHH_POS];
  ws[i] = (half_t)v;
}

// ---------------------------------------------------------------------------
// Fused TGN memory update. grid = (NN/TM, 2); blockIdx.y: 0 = mem, 1 = pos.
// ---------------------------------------------------------------------------
__global__ __launch_bounds__(256) void tgn_fused_wmma(
    const float* __restrict__ memory,  const float* __restrict__ pos_mem,
    const float* __restrict__ pos_emb,
    const float* __restrict__ raw_s,   const float* __restrict__ raw_d,
    const int*   __restrict__ other_s, const int*   __restrict__ other_d,
    const int*   __restrict__ t_s,     const int*   __restrict__ t_d,
    const int*   __restrict__ last_update,
    const float* __restrict__ w_t_mem, const float* __restrict__ b_t_mem,
    const float* __restrict__ w_t_pos, const float* __restrict__ b_t_pos,
    const half_t* __restrict__ wsh,
    const float* __restrict__ bih_mem, const float* __restrict__ bhh_mem,
    const float* __restrict__ bih_pos, const float* __restrict__ bhh_pos,
    float* __restrict__ out) {
  const int branch = blockIdx.y;
  const int m0     = blockIdx.x * TM;
  const int tid    = threadIdx.x;

  const float*  mem   = branch ? pos_mem  : memory;
  const float*  wt    = branch ? w_t_pos  : w_t_mem;
  const float*  bt    = branch ? b_t_pos  : b_t_mem;
  const float*  bih   = branch ? bih_pos  : bih_mem;
  const float*  bhh   = branch ? bhh_pos  : bhh_mem;
  const half_t* Wih_h = wsh + (branch ? OFF_WIH_POS : OFF_WIH_MEM);
  const half_t* Whh_h = wsh + (branch ? OFF_WHH_POS : OFF_WHH_MEM);

  __shared__ half_t sA[TM * SA_STRIDE];   // aggregated message tile, f16 (~66.6 KB)
  __shared__ int    sOther[TM];
  __shared__ int    sPick[TM];
  __shared__ int    sFeat[TM];            // pos-branch feature row
  __shared__ float  sDt[TM];

  // --- Phase 1a: per-row scalars (LastAggregator pick) ----------------------
  if (tid < TM) {
    int row  = m0 + tid;
    int ts   = t_s[row], td = t_d[row];
    int pick = (ts >= td) ? 1 : 0;
    sPick[tid]  = pick;
    sOther[tid] = pick ? other_s[row] : other_d[row];
    sFeat[tid]  = pick ? row : other_d[row];
    sDt[tid]    = (float)((pick ? ts : td) - last_update[row]);
    if (branch == 0) {  // new last_update = max(t_s, t_d)
      out[(size_t)2 * NN * D + row] = (float)(ts > td ? ts : td);
    }
  }
  __syncthreads();

  // --- Phase 1b: build aggr[TM,512] in f16 ---------------------------------
  // cols [0,128): h   [128,256): mem[other]   [256,384): feat   [384,512): time enc
  for (int it = tid; it < TM * (KIN / 4); it += 256) {
    int row = it >> 7;            // it / 128
    int c4  = (it & 127) << 2;    // column (multiple of 4)
    float4 v;
    if (c4 < D) {
      v = *(const float4*)(mem + (size_t)(m0 + row) * D + c4);
    } else if (c4 < 2 * D) {
      v = *(const float4*)(mem + (size_t)sOther[row] * D + (c4 - D));
    } else if (c4 < 3 * D) {
      int c = c4 - 2 * D;
      const float* f;
      if (branch == 0) f = (sPick[row] ? raw_s : raw_d) + (size_t)(m0 + row) * D + c;
      else             f = pos_emb + (size_t)sFeat[row] * D + c;
      v = *(const float4*)f;
    } else {
      int c = c4 - 3 * D;
      float dt = sDt[row];
      v.x = __cosf(fmaf(dt, wt[c + 0], bt[c + 0]));
      v.y = __cosf(fmaf(dt, wt[c + 1], bt[c + 1]));
      v.z = __cosf(fmaf(dt, wt[c + 2], bt[c + 2]));
      v.w = __cosf(fmaf(dt, wt[c + 3], bt[c + 3]));
    }
    half_t* dst = sA + row * SA_STRIDE + c4;
    dst[0] = (half_t)v.x; dst[1] = (half_t)v.y;
    dst[2] = (half_t)v.z; dst[3] = (half_t)v.w;
  }
  __syncthreads();

  // --- Phase 2: WMMA GEMMs -------------------------------------------------
  // Wave w owns output columns [16w, 16w+16).  GRU gate structure:
  //   r,z gates need gi+gh  -> accumulate Whh GEMM into the SAME accumulator.
  //   n gate needs gi_n and gh_n separately (n = tanh(inn + r*hn)).
  const int wave   = tid >> 5;
  const int lane   = tid & 31;
  const int laneN  = lane & 15;
  const int laneHi = lane >> 4;
  const int n0     = wave * 16;

  v8f acc_rz[2][NMT];   // fused gi+gh for r and z gates
  v8f acc_in[NMT];      // gi n-part
  v8f acc_hn[NMT];      // gh n-part
#pragma unroll
  for (int mt = 0; mt < NMT; ++mt) {
    acc_rz[0][mt] = (v8f)0.0f; acc_rz[1][mt] = (v8f)0.0f;
    acc_in[mt] = (v8f)0.0f;    acc_hn[mt] = (v8f)0.0f;
  }

  // r and z gates: aggr @ Wih^T then h @ Whh^T into the same accumulators
#pragma unroll
  for (int g = 0; g < 2; ++g) {
    const half_t* wbi = Wih_h + (size_t)(g * D + n0 + laneN) * KIN + laneHi * 16;
#pragma unroll 4
    for (int kt = 0; kt < 16; ++kt) {
      v16h B = load_bfrag(wbi + kt * 32);
#pragma unroll
      for (int mt = 0; mt < NMT; ++mt) {
        v16h A = load_afrag(sA + (mt * 16 + laneN) * SA_STRIDE, kt * 32, laneHi);
        acc_rz[g][mt] = __builtin_amdgcn_wmma_f32_16x16x32_f16(
            false, A, false, B, (short)0, acc_rz[g][mt], false, false);
      }
    }
    const half_t* wbh = Whh_h + (size_t)(g * D + n0 + laneN) * D + laneHi * 16;
#pragma unroll
    for (int kt = 0; kt < 4; ++kt) {
      v16h B = load_bfrag(wbh + kt * 32);
#pragma unroll
      for (int mt = 0; mt < NMT; ++mt) {
        v16h A = load_afrag(sA + (mt * 16 + laneN) * SA_STRIDE, kt * 32, laneHi);
        acc_rz[g][mt] = __builtin_amdgcn_wmma_f32_16x16x32_f16(
            false, A, false, B, (short)0, acc_rz[g][mt], false, false);
      }
    }
  }

  // n gate: keep input and hidden parts separate
  {
    const half_t* wbi = Wih_h + (size_t)(2 * D + n0 + laneN) * KIN + laneHi * 16;
#pragma unroll 4
    for (int kt = 0; kt < 16; ++kt) {
      v16h B = load_bfrag(wbi + kt * 32);
#pragma unroll
      for (int mt = 0; mt < NMT; ++mt) {
        v16h A = load_afrag(sA + (mt * 16 + laneN) * SA_STRIDE, kt * 32, laneHi);
        acc_in[mt] = __builtin_amdgcn_wmma_f32_16x16x32_f16(
            false, A, false, B, (short)0, acc_in[mt], false, false);
      }
    }
    const half_t* wbh = Whh_h + (size_t)(2 * D + n0 + laneN) * D + laneHi * 16;
#pragma unroll
    for (int kt = 0; kt < 4; ++kt) {
      v16h B = load_bfrag(wbh + kt * 32);
#pragma unroll
      for (int mt = 0; mt < NMT; ++mt) {
        v16h A = load_afrag(sA + (mt * 16 + laneN) * SA_STRIDE, kt * 32, laneHi);
        acc_hn[mt] = __builtin_amdgcn_wmma_f32_16x16x32_f16(
            false, A, false, B, (short)0, acc_hn[mt], false, false);
      }
    }
  }

  // --- Phase 3: GRU gates + output (all in registers) ----------------------
  const int j = n0 + laneN;           // output column of this lane
  const float b_r  = bih[j] + bhh[j];
  const float b_z  = bih[D + j] + bhh[D + j];
  const float bi_n = bih[2 * D + j];
  const float bh_n = bhh[2 * D + j];
  float* outz = out + (size_t)branch * NN * D;

#pragma unroll
  for (int mt = 0; mt < NMT; ++mt) {
#pragma unroll
    for (int r = 0; r < 8; ++r) {
      int row = m0 + mt * 16 + laneHi * 8 + r;   // C/D layout: M = r + 8*(lane>=16)
      float grv = acc_rz[0][mt][r] + b_r;
      float gzv = acc_rz[1][mt][r] + b_z;
      float gin = acc_in[mt][r] + bi_n;
      float ghn = acc_hn[mt][r] + bh_n;
      float rg = 1.0f / (1.0f + __expf(-grv));
      float zg = 1.0f / (1.0f + __expf(-gzv));
      float ng = tanhf(fmaf(rg, ghn, gin));
      float h  = mem[(size_t)row * D + j];       // hits L1/L2 (read in phase 1)
      outz[(size_t)row * D + j] = fmaf(zg, h - ng, ng);  // (1-z)*n + z*h
    }
  }
}

// ---------------------------------------------------------------------------
extern "C" void kernel_launch(void* const* d_in, const int* in_sizes, int n_in,
                              void* d_out, int out_size, void* d_ws, size_t ws_size,
                              hipStream_t stream) {
  (void)in_sizes; (void)n_in; (void)out_size; (void)ws_size;
  const float* memory  = (const float*)d_in[1];
  const float* pos_mem = (const float*)d_in[2];
  const float* pos_emb = (const float*)d_in[3];
  const float* raw_s   = (const float*)d_in[4];
  const float* raw_d   = (const float*)d_in[5];
  const int*   other_s = (const int*)d_in[6];
  const int*   other_d = (const int*)d_in[7];
  const int*   t_s     = (const int*)d_in[8];
  const int*   t_d     = (const int*)d_in[9];
  const int*   last_up = (const int*)d_in[10];
  const float* w_t_mem = (const float*)d_in[11];
  const float* b_t_mem = (const float*)d_in[12];
  const float* w_t_pos = (const float*)d_in[13];
  const float* b_t_pos = (const float*)d_in[14];
  const float* Wih_mem = (const float*)d_in[15];
  const float* Whh_mem = (const float*)d_in[16];
  const float* bih_mem = (const float*)d_in[17];
  const float* bhh_mem = (const float*)d_in[18];
  const float* Wih_pos = (const float*)d_in[19];
  const float* Whh_pos = (const float*)d_in[20];
  const float* bih_pos = (const float*)d_in[21];
  const float* bhh_pos = (const float*)d_in[22];

  half_t* wsh = (half_t*)d_ws;   // needs W_TOTAL*2 = 983,040 bytes of scratch

  convert_weights_f16<<<(W_TOTAL + 255) / 256, 256, 0, stream>>>(
      Wih_mem, Whh_mem, Wih_pos, Whh_pos, wsh);

  dim3 grid(NN / TM, 2);
  tgn_fused_wmma<<<grid, 256, 0, stream>>>(
      memory, pos_mem, pos_emb, raw_s, raw_d,
      other_s, other_d, t_s, t_d, last_up,
      w_t_mem, b_t_mem, w_t_pos, b_t_pos,
      wsh, bih_mem, bhh_mem, bih_pos, bhh_pos,
      (float*)d_out);
}
